// SL_CSC_IHT_26740466385538
// MI455X (gfx1250) — matrix operand order repaired and checked by reading, not compiled, not run on hardware
//
#include <hip/hip_runtime.h>
#include <hip/hip_bf16.h>
#include <stddef.h>

// ---------------- problem constants (from reference) ----------------
#define BATCH   32
#define IMG     64
#define NPIX    (IMG*IMG)          // 4096
#define NATOM   128
#define KATOM   8                  // 8x8 atoms
#define KSZ     (KATOM*KATOM)      // 64 = GEMM K
#define OSP     57                 // 64-8+1
#define NP      (OSP*OSP)          // 3249 positions per atom map
#define NPS     (NATOM*NP)         // 415872 per-sample X elements
#define NTOT    (BATCH*NPS)        // 13307904 total output elements
#define KSEL    256                // top-k per sample
#define TM_TILES 204               // ceil(3249/16)
#define T_SC    3

typedef float v2f  __attribute__((ext_vector_type(2)));
typedef float v8f  __attribute__((ext_vector_type(8)));

#define HAVE_WMMA_F32X4 __has_builtin(__builtin_amdgcn_wmma_f32_16x16x4_f32)
#if !HAVE_WMMA_F32X4
typedef __bf16 v16bf __attribute__((ext_vector_type(16)));
#endif

// ---------------- K0: zero dense X (in d_out) + control state ----------------
__global__ void init_kernel(float* __restrict__ X, int* __restrict__ xs_cnt,
                            int* __restrict__ ctrl_i) {
    size_t i = (size_t)blockIdx.x * blockDim.x + threadIdx.x;
    if (i < (size_t)NTOT) X[i] = 0.0f;
    if (i < BATCH) xs_cnt[i] = 0;
    if (i == 0) ctrl_i[0] = 0;
}

// ---------------- K1/K4: sparse transposed-conv reconstruction + error -------
// recon[b,y,x] = sum over sparse entries v at (a,h,w): v * W[a,0,y-h,x-w]
// err_s[b] = sum (Y - recon)^2 ; optionally store R = Y - recon.
__global__ void recon_kernel(const float* __restrict__ Y, const float* __restrict__ W,
                             const int* __restrict__ idxL, const float* __restrict__ valL,
                             const int* __restrict__ cntL,
                             float* __restrict__ Rout, float* __restrict__ err_s,
                             const int* __restrict__ ctrl_i, int check_acc) {
    if (check_acc && ctrl_i[0]) return;            // line search already satisfied
    const int b = blockIdx.x, tid = threadIdx.x;   // 256 threads
    __shared__ float img[NPIX];
    __shared__ float sf[256];
    for (int i = tid; i < NPIX; i += 256) img[i] = 0.0f;
    __syncthreads();
    const int cnt = cntL[b];
    for (int e = tid; e < cnt; e += 256) {
        const int   idx = idxL[b*KSEL + e];
        const float v   = valL[b*KSEL + e];
        const int a = idx / NP;
        const int p = idx - a*NP;
        const int h = p / OSP, w = p - h*OSP;
        const float* Wa = W + a*KSZ;
        #pragma unroll
        for (int i2 = 0; i2 < KATOM; ++i2)
            #pragma unroll
            for (int j2 = 0; j2 < KATOM; ++j2)
                atomicAdd(&img[(h+i2)*IMG + (w+j2)], v * Wa[i2*KATOM + j2]);
    }
    __syncthreads();
    const float* Yb = Y + b*NPIX;
    float ssq = 0.0f;
    for (int i = tid; i < NPIX; i += 256) {
        const float r = Yb[i] - img[i];
        if (Rout) Rout[b*NPIX + i] = r;
        ssq += r*r;
    }
    sf[tid] = ssq; __syncthreads();
    for (int s = 128; s > 0; s >>= 1) { if (tid < s) sf[tid] += sf[tid+s]; __syncthreads(); }
    if (tid == 0) err_s[b] = sf[0];
}

// ---------------- K1b: l2_start = sum(err_s); reset accepted -----------------
__global__ void l2_kernel(const float* __restrict__ err_s, float* __restrict__ ctrl_f,
                          int* __restrict__ ctrl_i) {
    __shared__ float sf[BATCH];
    const int tid = threadIdx.x;
    sf[tid] = err_s[tid]; __syncthreads();
    for (int s = 16; s > 0; s >>= 1) { if (tid < s) sf[tid] += sf[tid+s]; __syncthreads(); }
    if (tid == 0) { ctrl_f[0] = sf[0]; ctrl_i[0] = 0; }
}

// ---------------- K2: g = conv_Dt(R, Wt) via V_WMMA_F32_16X16X4_F32 ----------
// GEMM per sample: [3249 pos x 64 patch] x [64 x 128 atoms], fp32 throughout.
// One wave per 16x16 tile; 8 waves of a block share one position tile.
// Wt[a,0,i,j] = W[a,0,j,i]  ->  B(k,a) = W[a*64 + (k%8)*8 + (k/8)], k = i*8+j.
__global__ void convdt_wmma_kernel(const float* __restrict__ R, const float* __restrict__ W,
                                   float* __restrict__ G) {
    const int lane = threadIdx.x & 31;
    const int wv   = threadIdx.x >> 5;          // atom tile 0..7
    const int tm   = blockIdx.x % TM_TILES;     // position tile
    const int b    = blockIdx.x / TM_TILES;
    const int lo   = lane & 15;
    const int grp  = lane >> 4;

    // A fragment source: row M = position (clamped on the last ragged tile)
    int p = tm*16 + lo; if (p > NP-1) p = NP-1;
    const int h = p / OSP, w = p - (p/OSP)*OSP;
    const float* Rp = R + b*NPIX + h*IMG + w;

    // B fragment source: column N = atom (same lane->N map as D)
    const int a = wv*16 + lo;
    const float* Wa = W + a*KSZ;

    v8f acc = {0.f,0.f,0.f,0.f,0.f,0.f,0.f,0.f};

#if HAVE_WMMA_F32X4
    #pragma unroll
    for (int ks = 0; ks < 16; ++ks) {           // K = 64 in steps of 4
        const int k0 = ks*4 + grp*2;            // lanes 0-15: k0,k0+1 ; 16-31: k0+2,k0+3
        const int k1 = k0 + 1;
        v2f av, bv;
        av.x = Rp[(k0 >> 3)*IMG + (k0 & 7)];
        av.y = Rp[(k1 >> 3)*IMG + (k1 & 7)];
        bv.x = Wa[(k0 & 7)*KATOM + (k0 >> 3)];
        bv.y = Wa[(k1 & 7)*KATOM + (k1 >> 3)];
        acc = __builtin_amdgcn_wmma_f32_16x16x4_f32(
                  false, av, false, bv, (short)0, acc, false, false);
    }
#else
    // bf16 fallback (codegen-verified shape): two 16x16x32 steps cover K=64.
    #pragma unroll
    for (int kc = 0; kc < 2; ++kc) {
        const int kb0 = kc*32;
        v16bf av, bv;
        #pragma unroll
        for (int e = 0; e < 16; ++e) {
            const int ka = kb0 + ((e < 8) ? (e + 8*grp) : (8 + e + 8*grp));
            av[e] = (__bf16)Rp[(ka >> 3)*IMG + (ka & 7)];
            const int kb = kb0 + ((e < 8) ? (e + 8*grp) : (8 + e + 8*grp));
            bv[e] = (__bf16)Wa[(kb & 7)*KATOM + (kb >> 3)];
        }
        acc = __builtin_amdgcn_wmma_f32_16x16x32_bf16(
                  false, av, false, bv, (short)0, acc, false, false);
    }
#endif

    // D layout: VGPR r, lanes 0-15 -> (M=r, N=lane); lanes 16-31 -> (M=r+8, N=lane-16)
    const int pbase = tm*16 + grp*8;
    float* Gp = G + (size_t)b*NPS + (size_t)a*NP;
    #pragma unroll
    for (int r = 0; r < 8; ++r) {
        const int pp = pbase + r;
        if (pp < NP) Gp[pp] = acc[r];
    }
}

// ---------------- K3: top-256 of |X + alpha*G| per sample --------------------
__global__ void topk_kernel(const float* __restrict__ X, const float* __restrict__ G,
                            float alpha, int* __restrict__ cd_idx, float* __restrict__ cd_val,
                            int* __restrict__ cd_cnt, const int* __restrict__ ctrl_i) {
    if (ctrl_i[0]) return;
    const int b = blockIdx.x, tid = threadIdx.x;   // 512 threads
    const float* Xb = X + (size_t)b*NPS;
    const float* Gb = G + (size_t)b*NPS;
    __shared__ float sf[512];
    __shared__ int   si[512];
    __shared__ float s_lo, s_hi;
    __shared__ int   s_n;

    float m = 0.0f;
    for (int i = tid; i < NPS; i += 512)
        m = fmaxf(m, fabsf(Xb[i] + alpha*Gb[i]));
    sf[tid] = m; __syncthreads();
    for (int s = 256; s > 0; s >>= 1) { if (tid < s) sf[tid] = fmaxf(sf[tid], sf[tid+s]); __syncthreads(); }
    if (tid == 0) { s_hi = sf[0]; s_lo = 0.0f; }
    __syncthreads();

    // bisection on threshold t: keep count(|z| >= s_lo) >= KSEL
    for (int it = 0; it < 32; ++it) {
        const float mid = 0.5f*(s_lo + s_hi);
        int c = 0;
        for (int i = tid; i < NPS; i += 512)
            c += (fabsf(Xb[i] + alpha*Gb[i]) >= mid) ? 1 : 0;
        si[tid] = c; __syncthreads();
        for (int s = 256; s > 0; s >>= 1) { if (tid < s) si[tid] += si[tid+s]; __syncthreads(); }
        if (tid == 0) { if (si[0] >= KSEL) s_lo = mid; else s_hi = mid; }
        __syncthreads();
    }
    const float thr = s_lo;
    if (tid == 0) s_n = 0;
    __syncthreads();
    for (int i = tid; i < NPS; i += 512) {
        const float z = Xb[i] + alpha*Gb[i];
        if (fabsf(z) >= thr) {
            const int slot = atomicAdd(&s_n, 1);
            if (slot < KSEL) { cd_idx[b*KSEL + slot] = i; cd_val[b*KSEL + slot] = z; }
        }
    }
    __syncthreads();
    if (tid == 0) cd_cnt[b] = (s_n < KSEL) ? s_n : KSEL;
}

// ---------------- K5: accept if total err < l2_start (or last attempt) -------
__global__ void decide_kernel(const float* __restrict__ err_s, const float* __restrict__ ctrl_f,
                              int* __restrict__ ctrl_i, int is_last) {
    __shared__ float sf[BATCH];
    const int tid = threadIdx.x;
    sf[tid] = err_s[tid]; __syncthreads();
    for (int s = 16; s > 0; s >>= 1) { if (tid < s) sf[tid] += sf[tid+s]; __syncthreads(); }
    if (tid == 0) {
        if (!ctrl_i[0] && (sf[0] < ctrl_f[0] || is_last)) ctrl_i[0] = 1;
    }
}

// ---------------- K6: commit candidate -> X (sparse list + dense mirror) -----
__global__ void commit_kernel(float* __restrict__ X,
                              int* __restrict__ xs_idx, float* __restrict__ xs_val,
                              int* __restrict__ xs_cnt,
                              const int* __restrict__ cd_idx, const float* __restrict__ cd_val,
                              const int* __restrict__ cd_cnt) {
    const int b = blockIdx.x, tid = threadIdx.x;   // 256 threads
    const int oldc = xs_cnt[b];
    for (int e = tid; e < oldc; e += 256)
        X[(size_t)b*NPS + xs_idx[b*KSEL + e]] = 0.0f;
    __syncthreads();
    const int newc = cd_cnt[b];
    for (int e = tid; e < newc; e += 256) {
        const int   id = cd_idx[b*KSEL + e];
        const float v  = cd_val[b*KSEL + e];
        xs_idx[b*KSEL + e] = id;
        xs_val[b*KSEL + e] = v;
        X[(size_t)b*NPS + id] = v;
    }
    if (tid == 0) xs_cnt[b] = newc;
}

// ---------------- host-side launch sequence ----------------------------------
extern "C" void kernel_launch(void* const* d_in, const int* in_sizes, int n_in,
                              void* d_out, int out_size, void* d_ws, size_t ws_size,
                              hipStream_t stream) {
    (void)in_sizes; (void)n_in; (void)out_size; (void)ws_size;
    const float* Y = (const float*)d_in[0];   // (32,1,64,64)
    const float* W = (const float*)d_in[1];   // (128,1,8,8) already L2-normalized
    float* X = (float*)d_out;                 // dense X lives in d_out (final answer)

    // workspace carve-up (~54 MB; G dominates and stays L2-resident)
    char* ws = (char*)d_ws;
    size_t off = 0;
    auto carve = [&](size_t bytes) { void* p = ws + off; off = (off + bytes + 255) & ~(size_t)255; return p; };
    float* G      = (float*)carve((size_t)NTOT * 4);        // gradient, dense
    float* Rbuf   = (float*)carve((size_t)BATCH * NPIX * 4);
    int*   xs_idx = (int*)  carve(BATCH * KSEL * 4);
    float* xs_val = (float*)carve(BATCH * KSEL * 4);
    int*   xs_cnt = (int*)  carve(BATCH * 4);
    int*   cd_idx = (int*)  carve(BATCH * KSEL * 4);
    float* cd_val = (float*)carve(BATCH * KSEL * 4);
    int*   cd_cnt = (int*)  carve(BATCH * 4);
    float* err_s  = (float*)carve(BATCH * 4);
    float* ctrl_f = (float*)carve(64);                       // [0] = l2_start
    int*   ctrl_i = (int*)  carve(64);                       // [0] = accepted flag

    const int zgrid = (NTOT + 255) / 256;
    hipLaunchKernelGGL(init_kernel, dim3(zgrid), dim3(256), 0, stream, X, xs_cnt, ctrl_i);

    for (int it = 0; it < T_SC; ++it) {
        // R = Y - convT(X,W), l2_start = ||R||^2, accepted = 0
        hipLaunchKernelGGL(recon_kernel, dim3(BATCH), dim3(256), 0, stream,
                           Y, W, xs_idx, xs_val, xs_cnt, Rbuf, err_s, ctrl_i, 0);
        hipLaunchKernelGGL(l2_kernel, dim3(1), dim3(BATCH), 0, stream, err_s, ctrl_f, ctrl_i);
        // g = conv(R, Wt)  — the WMMA GEMM
        hipLaunchKernelGGL(convdt_wmma_kernel, dim3(BATCH * TM_TILES), dim3(256), 0, stream,
                           Rbuf, W, G);
        // line search: alpha = 1, 1/2, ..., 2^-16; predicated on accept flag
        for (int t = 0; t <= 16; ++t) {
            const float alpha = 1.0f / (float)(1u << t);
            hipLaunchKernelGGL(topk_kernel, dim3(BATCH), dim3(512), 0, stream,
                               X, G, alpha, cd_idx, cd_val, cd_cnt, ctrl_i);
            hipLaunchKernelGGL(recon_kernel, dim3(BATCH), dim3(256), 0, stream,
                               Y, W, cd_idx, cd_val, cd_cnt, (float*)nullptr, err_s, ctrl_i, 1);
            hipLaunchKernelGGL(decide_kernel, dim3(1), dim3(BATCH), 0, stream,
                               err_s, ctrl_f, ctrl_i, (t == 16) ? 1 : 0);
        }
        hipLaunchKernelGGL(commit_kernel, dim3(BATCH), dim3(256), 0, stream,
                           X, xs_idx, xs_val, xs_cnt, cd_idx, cd_val, cd_cnt);
    }
}
